// InceptionFeatureExtractor_69922067579634
// MI455X (gfx1250) — matrix operand order repaired and checked by reading, not compiled, not run on hardware
//
#include <hip/hip_runtime.h>
#include <hip/hip_bf16.h>
#include <stdint.h>

typedef __attribute__((ext_vector_type(16))) _Float16 v16h;
typedef __attribute__((ext_vector_type(8)))  float    v8f;

#define NPTS 16384
#define CH   64

// ---------------- WMMA helpers (CDNA5 v_wmma_f32_16x16x32_f16) ----------------

static __device__ inline v8f wmma_f16(v16h a, v16h b, v8f c) {
  // 8 args: (neg_a, A, neg_b, B, c_mod, C, reuse_a, reuse_b)
  return __builtin_amdgcn_wmma_f32_16x16x32_f16(false, a, false, b, (short)0, c, false, false);
}

// A fragment (16x32, f16) from f32 row-major source.
// Layout: lanes 0-15: M=lane, VGPR v holds K=2v,2v+1;
//         lanes 16-31: M=lane-16, +16 on K.
static __device__ inline v16h load_a_frag_f32(const float* src, int ld, int r0, int kb, int lane) {
  int m = lane & 15, hi = lane >> 4;
  const float* row = src + (size_t)(r0 + m) * ld;
  v16h a;
#pragma unroll
  for (int v = 0; v < 8; ++v) {
    int k = kb + 2 * v + hi * 16;
    float2 f = *(const float2*)(row + k);
    a[2 * v]     = (_Float16)f.x;
    a[2 * v + 1] = (_Float16)f.y;
  }
  return a;
}

// B fragment (32x16, f16), B[k][n] = src[n0+n][k] (n-major source, f32).
// Layout: lanes 0-15: N=lane, VGPR v holds K=2v,2v+1; lanes 16-31: K += 16.
static __device__ inline v16h load_b_frag_f32(const float* src, int ld, int n0, int kb, int lane) {
  int n = lane & 15, hi = lane >> 4;
  const float* row = src + (size_t)(n0 + n) * ld;
  v16h b;
#pragma unroll
  for (int v = 0; v < 8; ++v) {
    int k = kb + 2 * v + hi * 16;
    float2 f = *(const float2*)(row + k);
    b[2 * v]     = (_Float16)f.x;
    b[2 * v + 1] = (_Float16)f.y;
  }
  return b;
}

// B fragment from pre-packed f16 n-major weights (Wt[64][Cin]); pairs are contiguous dwords.
static __device__ inline v16h load_b_frag_h16(const _Float16* w, int ld, int n0, int kb, int lane) {
  int n = lane & 15, hi = lane >> 4;
  const _Float16* row = w + (size_t)(n0 + n) * ld + kb + hi * 16;
  union { v16h v; unsigned u[8]; } b;
#pragma unroll
  for (int i = 0; i < 8; ++i) b.u[i] = *(const unsigned*)(row + 2 * i);
  return b.v;
}

// ---------------- Stage 0: kNN(k=1) on coords + pre EdgeConv ----------------

__global__ __launch_bounds__(256) void pre_kernel(const float* __restrict__ x,
                                                  const float* __restrict__ Wp,
                                                  const float* __restrict__ bp,
                                                  float* __restrict__ h,
                                                  float* __restrict__ res) {
  int w = threadIdx.x >> 5, lane = threadIdx.x & 31;
  int i = blockIdx.x * 8 + w;
  float xi0 = x[(size_t)i * 3], xi1 = x[(size_t)i * 3 + 1], xi2 = x[(size_t)i * 3 + 2];
  float bestd = 3.4e38f; int bestj = 0;
  for (int j = lane; j < NPTS; j += 32) {
    float d0 = x[(size_t)j * 3] - xi0;
    float d1 = x[(size_t)j * 3 + 1] - xi1;
    float d2 = x[(size_t)j * 3 + 2] - xi2;
    float d = d0 * d0 + d1 * d1 + d2 * d2;
    if (j != i && (d < bestd || (d == bestd && j < bestj))) { bestd = d; bestj = j; }
  }
#pragma unroll
  for (int off = 16; off; off >>= 1) {
    float od = __shfl_xor(bestd, off, 32);
    int   oj = __shfl_xor(bestj, off, 32);
    if (od < bestd || (od == bestd && oj < bestj)) { bestd = od; bestj = oj; }
  }
  float xj0 = x[(size_t)bestj * 3] - xi0;
  float xj1 = x[(size_t)bestj * 3 + 1] - xi1;
  float xj2 = x[(size_t)bestj * 3 + 2] - xi2;
  for (int c = lane; c < CH; c += 32) {
    float acc = bp[c] + xi0 * Wp[0 * CH + c] + xi1 * Wp[1 * CH + c] + xi2 * Wp[2 * CH + c]
                      + xj0 * Wp[3 * CH + c] + xj1 * Wp[4 * CH + c] + xj2 * Wp[5 * CH + c];
    h[(size_t)i * CH + c] = fmaxf(acc, 0.0f);
    res[(size_t)i * CH + c] = 0.0f;
  }
}

// ---------------- Row norms + copy h into X[:, :64] ----------------

__global__ __launch_bounds__(256) void rownorm_kernel(const float* __restrict__ h,
                                                      float* __restrict__ s,
                                                      float* __restrict__ X) {
  int w = threadIdx.x >> 5, lane = threadIdx.x & 31;
  int n = blockIdx.x * 8 + w;
  float v0 = h[(size_t)n * CH + lane];
  float v1 = h[(size_t)n * CH + 32 + lane];
  X[(size_t)n * 256 + lane] = v0;
  X[(size_t)n * 256 + 32 + lane] = v1;
  float sum = v0 * v0 + v1 * v1;
#pragma unroll
  for (int off = 16; off; off >>= 1) sum += __shfl_xor(sum, off, 32);
  if (lane == 0) s[n] = sum;
}

// ---------------- kNN: WMMA Gram tile + per-chunk bitonic top-40 ----------------
// Key pack: high 18 bits of non-negative f32 distance | 14-bit column index.

__global__ __launch_bounds__(256) void knn_chunk_kernel(const float* __restrict__ h,
                                                        const float* __restrict__ s,
                                                        unsigned* __restrict__ cand) {
  __shared__ unsigned keys[16][1024];  // 64 KB
  int bx = blockIdx.x;
  int r0 = (bx >> 4) << 4;       // 16-row tile
  int chunk = bx & 15;
  int c0 = chunk << 10;          // 1024-column chunk
  int tid = threadIdx.x;
  int w = tid >> 5, lane = tid & 31;
  int hi = lane >> 4;

  v16h a0 = load_a_frag_f32(h, CH, r0, 0, lane);
  v16h a1 = load_a_frag_f32(h, CH, r0, 32, lane);
  float sr[8];
#pragma unroll
  for (int r = 0; r < 8; ++r) sr[r] = s[r0 + r + hi * 8];

  for (int st = w; st < 64; st += 8) {
    int cb = c0 + st * 16;
    v16h b0 = load_b_frag_f32(h, CH, cb, 0, lane);
    v16h b1 = load_b_frag_f32(h, CH, cb, 32, lane);
    v8f acc = {};
    acc = wmma_f16(a0, b0, acc);
    acc = wmma_f16(a1, b1, acc);
    int nloc = lane & 15;
    int gcol = cb + nloc;
    float sc = s[gcol];
#pragma unroll
    for (int r = 0; r < 8; ++r) {
      int M = r + hi * 8;
      int grow = r0 + M;
      // branchless: pack (18-bit dist | 14-bit idx); force to UINT_MAX for self
      float d = fmaxf(sr[r] + sc - 2.0f * acc[r], 0.0f);
      unsigned key = (__float_as_uint(d) & 0xFFFFC000u) | (unsigned)gcol;
      key |= (unsigned)(-(int)(gcol == grow));
      keys[M][st * 16 + nloc] = key;
    }
  }
  __syncthreads();

  // Batched bitonic sort: 16 rows x 1024 keys, ascending.
  for (int size = 2; size <= 1024; size <<= 1) {
    for (int stride = size >> 1; stride > 0; stride >>= 1) {
      for (int e = tid; e < 16 * 512; e += 256) {
        int row = e >> 9, i = e & 511;
        int l = ((i & ~(stride - 1)) << 1) | (i & (stride - 1));
        int pr = l + stride;
        bool asc = ((l & size) == 0);
        unsigned x0 = keys[row][l], x1 = keys[row][pr];
        if ((x0 > x1) == asc) { keys[row][l] = x1; keys[row][pr] = x0; }
      }
      __syncthreads();
    }
  }
  for (int e = tid; e < 16 * 40; e += 256) {
    int row = e / 40, j = e % 40;
    cand[(size_t)(r0 + row) * 640 + chunk * 40 + j] = keys[row][j];
  }
}

__global__ __launch_bounds__(256) void knn_merge_kernel(const unsigned* __restrict__ cand,
                                                        int* __restrict__ nbr) {
  __shared__ unsigned buf[1024];
  int n = blockIdx.x, tid = threadIdx.x;
  for (int e = tid; e < 1024; e += 256)
    buf[e] = (e < 640) ? cand[(size_t)n * 640 + e] : 0xFFFFFFFFu;
  __syncthreads();
  for (int size = 2; size <= 1024; size <<= 1) {
    for (int stride = size >> 1; stride > 0; stride >>= 1) {
      for (int e = tid; e < 512; e += 256) {
        int l = ((e & ~(stride - 1)) << 1) | (e & (stride - 1));
        int pr = l + stride;
        bool asc = ((l & size) == 0);
        unsigned x0 = buf[l], x1 = buf[pr];
        if ((x0 > x1) == asc) { buf[l] = x1; buf[pr] = x0; }
      }
      __syncthreads();
    }
  }
  if (tid < 40) nbr[(size_t)n * 40 + tid] = (int)(buf[tid] & 0x3FFFu);
}

// ---------------- EdgeConv as dense GEMMs + gather/max epilogue ----------------

// At[n][k] = W[k][n] - W[Cin+k][n]; Bt[n][k] = W[Cin+k][n]   (f16, n-major)
__global__ __launch_bounds__(256) void prep_w_kernel(const float* __restrict__ W, int Cin,
                                                     _Float16* __restrict__ At,
                                                     _Float16* __restrict__ Bt) {
  int t = blockIdx.x * 256 + threadIdx.x;
  if (t >= 64 * Cin) return;
  int n = t / Cin, k = t % Cin;
  float top = W[(size_t)k * CH + n];
  float bot = W[(size_t)(Cin + k) * CH + n];
  At[(size_t)n * Cin + k] = (_Float16)(top - bot);
  Bt[(size_t)n * Cin + k] = (_Float16)bot;
}

// out[N][64] = X[:, :Cin] @ W  (W given n-major f16). One 16x16 tile per wave.
__global__ __launch_bounds__(256) void gemm_kernel(const float* __restrict__ X, int ldx, int Cin,
                                                   const _Float16* __restrict__ Wt,
                                                   float* __restrict__ out) {
  int gw = blockIdx.x * 8 + (threadIdx.x >> 5);
  int lane = threadIdx.x & 31;
  int r0 = (gw >> 2) << 4;
  int n0 = (gw & 3) << 4;
  v8f acc = {};
  for (int kb = 0; kb < Cin; kb += 32) {
    v16h a = load_a_frag_f32(X, ldx, r0, kb, lane);
    v16h b = load_b_frag_h16(Wt, Cin, n0, kb, lane);
    acc = wmma_f16(a, b, acc);
  }
  int hi = lane >> 4;
#pragma unroll
  for (int r = 0; r < 8; ++r)
    out[(size_t)(r0 + r + hi * 8) * CH + n0 + (lane & 15)] = acc[r];
}

// X[n][col0+c] = relu(P[n][c] + max_k Q[nbr[n][k*stride]][c] + bias[c])
__global__ __launch_bounds__(256) void aggregate_kernel(const float* __restrict__ P,
                                                        const float* __restrict__ Q,
                                                        const int* __restrict__ nbr, int stride,
                                                        const float* __restrict__ bias,
                                                        float* __restrict__ X, int col0) {
  __shared__ int nb[4][20];
  int n0 = blockIdx.x * 4;
  int tid = threadIdx.x;
  if (tid < 80) {
    int r = tid / 20, k = tid % 20;
    nb[r][k] = nbr[(size_t)(n0 + r) * 40 + k * stride];
  }
  __syncthreads();
  int r = tid >> 6, c = tid & 63;
  int n = n0 + r;
  float best = -3.4e38f;
#pragma unroll 4
  for (int k = 0; k < 20; ++k) best = fmaxf(best, Q[(size_t)nb[r][k] * CH + c]);
  float v = fmaxf(P[(size_t)n * CH + c] + best + bias[c], 0.0f);
  X[(size_t)n * 256 + col0 + c] = v;
}

// reference reshape(N, 64, 4).max(-1): max over 4 consecutive concat channels
__global__ __launch_bounds__(256) void groupmax_kernel(const float* __restrict__ X,
                                                       float* __restrict__ out) {
  int t = blockIdx.x * 256 + threadIdx.x;  // N*64
  int n = t >> 6, c = t & 63;
  const float* row = X + (size_t)n * 256 + 4 * c;
  out[t] = fmaxf(fmaxf(row[0], row[1]), fmaxf(row[2], row[3]));
}

__global__ __launch_bounds__(256) void combine_kernel(const float* __restrict__ B0,
                                                      const float* __restrict__ B1,
                                                      float* __restrict__ h,
                                                      float* __restrict__ res) {
  int t = blockIdx.x * 256 + threadIdx.x;
  float hn = fmaxf(B0[t], B1[t]) + h[t];
  h[t] = hn;
  res[t] += hn;
}

__global__ __launch_bounds__(256) void finalize_kernel(const float* __restrict__ res,
                                                       float* __restrict__ out) {
  int t = blockIdx.x * 256 + threadIdx.x;
  out[t] = res[t] * 0.5f;
}

// ---------------- Host orchestration ----------------

extern "C" void kernel_launch(void* const* d_in, const int* in_sizes, int n_in,
                              void* d_out, int out_size, void* d_ws, size_t ws_size,
                              hipStream_t stream) {
  (void)in_sizes; (void)n_in; (void)out_size; (void)ws_size;
  const float* x  = (const float*)d_in[0];
  const float* Wp = (const float*)d_in[1];
  const float* bp = (const float*)d_in[2];

  char* p = (char*)d_ws;
  auto alloc = [&](size_t bytes) { char* r = p; p += (bytes + 255) & ~(size_t)255; return r; };
  float*    h    = (float*)alloc((size_t)NPTS * CH * 4);
  float*    res  = (float*)alloc((size_t)NPTS * CH * 4);
  float*    X    = (float*)alloc((size_t)NPTS * 256 * 4);
  float*    s    = (float*)alloc((size_t)NPTS * 4);
  float*    P    = (float*)alloc((size_t)NPTS * CH * 4);
  float*    Q    = (float*)alloc((size_t)NPTS * CH * 4);
  float*    Bb0  = (float*)alloc((size_t)NPTS * CH * 4);
  float*    Bb1  = (float*)alloc((size_t)NPTS * CH * 4);
  unsigned* cand = (unsigned*)alloc((size_t)NPTS * 16 * 40 * 4);
  int*      nbr  = (int*)alloc((size_t)NPTS * 40 * 4);
  _Float16* At   = (_Float16*)alloc((size_t)64 * 192 * 2);
  _Float16* Bt   = (_Float16*)alloc((size_t)64 * 192 * 2);

  pre_kernel<<<NPTS / 8, 256, 0, stream>>>(x, Wp, bp, h, res);

  for (int l = 0; l < 2; ++l) {
    rownorm_kernel<<<NPTS / 8, 256, 0, stream>>>(h, s, X);
    knn_chunk_kernel<<<(NPTS / 16) * 16, 256, 0, stream>>>(h, s, cand);
    knn_merge_kernel<<<NPTS, 256, 0, stream>>>(cand, nbr);
    for (int g = 0; g < 2; ++g) {
      for (int t = 0; t < 3; ++t) {
        int Cin = CH * (t + 1);
        const float* W  = (const float*)d_in[3 + l * 6 + g * 3 + t];
        const float* bb = (const float*)d_in[15 + l * 6 + g * 3 + t];
        prep_w_kernel<<<(64 * Cin + 255) / 256, 256, 0, stream>>>(W, Cin, At, Bt);
        gemm_kernel<<<512, 256, 0, stream>>>(X, 256, Cin, At, P);
        gemm_kernel<<<512, 256, 0, stream>>>(X, 256, Cin, Bt, Q);
        aggregate_kernel<<<NPTS / 4, 256, 0, stream>>>(P, Q, nbr, g + 1, bb, X, CH * (t + 1));
      }
      groupmax_kernel<<<NPTS * CH / 256, 256, 0, stream>>>(X, g == 0 ? Bb0 : Bb1);
    }
    combine_kernel<<<NPTS * CH / 256, 256, 0, stream>>>(Bb0, Bb1, h, res);
  }
  finalize_kernel<<<NPTS * CH / 256, 256, 0, stream>>>(res, (float*)d_out);
}
